// DeepModel_multi_12945031430869
// MI455X (gfx1250) — matrix operand order repaired and checked by reading, not compiled, not run on hardware
//
#include <hip/hip_runtime.h>

// Problem constants (from the reference)
#define N_ROWS 4096
#define D_IN   512
#define D_H    2048
#define USED   32
#define OUT_ROW (USED * (D_IN + 1))   // 16416 floats per output row
#define NCH    (D_H / 64)             // 32 column chunks of 64

typedef __attribute__((ext_vector_type(16))) __bf16          v16bf;
typedef __attribute__((ext_vector_type(8)))  float           v8f;

__device__ __forceinline__ unsigned short f2bf(float f) {
  // round-to-nearest-even f32 -> bf16
  unsigned int u = __float_as_uint(f);
  u += 0x7FFFu + ((u >> 16) & 1u);
  return (unsigned short)(u >> 16);
}

struct B32x8 { uint4 lo, hi; };  // 32 bytes = one v16bf fragment

// Fused per-head GEMM (bf16 WMMA, f32 accum) + bias + relu + dot(W2) + b2.
// Block = 256 threads (8 waves) = 64 rows of x for one head, loops full D_H
// in 64-column chunks. Both x and W1 tiles staged in LDS as bf16; each wave
// owns one 16-row group and two 16-col tiles (2 accumulators per A fragment).
__global__ __launch_bounds__(256)
void fused_head_gemm(const float* __restrict__ x,
                     const float* __restrict__ W1,
                     const float* __restrict__ b1,
                     const float* __restrict__ W2,
                     const float* __restrict__ b2,
                     float* __restrict__ out) {
  // stride 520 bf16 = 1040B rows: 16B-aligned for ds_load_b128, 4-bank row step
  __shared__ unsigned short xs[64][520];   // x tile,  bf16 row-major   (~65 KB)
  __shared__ unsigned short ws[64][520];   // W1 chunk, bf16 transposed (~65 KB)
  __shared__ float ybuf[8][16];

  const int h    = blockIdx.y;          // head 0..31
  const int nb   = blockIdx.x * 64;     // row block
  const int tid  = threadIdx.x;
  const int lane = tid & 31;
  const int w    = tid >> 5;            // wave 0..7
  const int rg   = w & 3;               // row-group (16 rows each)
  const int ct   = w >> 2;              // which 32-col half of the 64-col chunk
  const int l16  = lane & 15;
  const int khalf = lane >> 4;          // A and B: K+16 half

  // ---- stage x tile (64 rows x 512) -> xs as bf16, packed b32 stores ----
  {
    const float* xb = x + (size_t)nb * D_IN;
    for (int i = tid; i < 64 * 256; i += 256) {       // 64 float2 per thread
      const int row = i >> 8;
      const int c2  = (i & 255) * 2;
      float2 v = *(const float2*)(xb + (size_t)row * D_IN + c2);
      unsigned int p = (unsigned int)f2bf(v.x) | ((unsigned int)f2bf(v.y) << 16);
      *(unsigned int*)&xs[row][c2] = p;
    }
  }
  // (made visible by the post-staging barrier inside the first chunk iteration)

  float yacc[8] = {0.f,0.f,0.f,0.f,0.f,0.f,0.f,0.f};

  // W1 staging decomposition: 32 threads x 2 cols per k-row, 8 k-rows per pass
  const int scol = (tid & 31) * 2;   // f_local 0..62
  const int srow = tid >> 5;         // k start 0..7, step 8

  // per-wave LDS fragment base pointers
  const unsigned short* aw = &xs[rg * 16 + l16][khalf * 16];
  const unsigned short* bw0 = &ws[ct * 32 + l16][khalf * 16];
  const unsigned short* bw1 = &ws[ct * 32 + 16 + l16][khalf * 16];

  for (int ft = 0; ft < NCH; ++ft) {
    const int f0g = ft * 64;         // global f base of this 64-col chunk
    __syncthreads();                 // protect ws/xs from previous readers
    // ---- stage W1[k=0..511, f0g..f0g+63] -> ws[f_local][k] as bf16 ----
    const float* wp = W1 + (size_t)h * D_IN * D_H + f0g + scol;
#pragma unroll 4
    for (int kk = srow; kk < D_IN; kk += 8) {
      float2 v = *(const float2*)(wp + (size_t)kk * D_H);
      // prefetch next chunk's row segment (4 threads/row, 64B apart -> 256B)
      if (((tid & 7) == 0) && (ft + 1 < NCH))
        __builtin_prefetch(wp + (size_t)kk * D_H + 64, 0, 3);
      ws[scol][kk]     = f2bf(v.x);
      ws[scol + 1][kk] = f2bf(v.y);
    }
    __syncthreads();

    // ---- two 16x16 WMMA tiles per wave, sharing each A fragment ----
    v8f acc0 = {0.f,0.f,0.f,0.f,0.f,0.f,0.f,0.f};
    v8f acc1 = {0.f,0.f,0.f,0.f,0.f,0.f,0.f,0.f};
#pragma unroll
    for (int kt = 0; kt < 16; ++kt) {
      B32x8 ta, tb0, tb1;
      ta.lo  = *(const uint4*)(aw  + kt * 32);
      ta.hi  = *(const uint4*)(aw  + kt * 32 + 8);
      tb0.lo = *(const uint4*)(bw0 + kt * 32);
      tb0.hi = *(const uint4*)(bw0 + kt * 32 + 8);
      tb1.lo = *(const uint4*)(bw1 + kt * 32);
      tb1.hi = *(const uint4*)(bw1 + kt * 32 + 8);
      v16bf af = __builtin_bit_cast(v16bf, ta);
      acc0 = __builtin_amdgcn_wmma_f32_16x16x32_bf16(
          false, af, false, __builtin_bit_cast(v16bf, tb0),
          (short)0, acc0, false, false);
      acc1 = __builtin_amdgcn_wmma_f32_16x16x32_bf16(
          false, af, false, __builtin_bit_cast(v16bf, tb1),
          (short)0, acc1, false, false);
    }

    // ---- fused epilogue: +b1, relu, * W2, accumulate y ----
    const int fcol0 = f0g + ct * 32 + l16;           // C/D: col = lane%16
    const float b1v0 = b1[(size_t)h * D_H + fcol0];
    const float w2v0 = W2[(size_t)h * D_H + fcol0];
    const float b1v1 = b1[(size_t)h * D_H + fcol0 + 16];
    const float w2v1 = W2[(size_t)h * D_H + fcol0 + 16];
#pragma unroll
    for (int r = 0; r < 8; ++r) {                    // C/D: row = r + 8*(lane/16)
      float v0 = fmaxf(acc0[r] + b1v0, 0.f);
      float v1 = fmaxf(acc1[r] + b1v1, 0.f);
      yacc[r] += v0 * w2v0 + v1 * w2v1;
    }
  }

  // ---- reduce y over the 16 lanes of each half (columns), combine halves ----
#pragma unroll
  for (int r = 0; r < 8; ++r) {
    float v = yacc[r];
    v += __shfl_xor(v, 1, 32);
    v += __shfl_xor(v, 2, 32);
    v += __shfl_xor(v, 4, 32);
    v += __shfl_xor(v, 8, 32);
    if (l16 == 0) ybuf[w][r + khalf * 8] = v;        // local row 0..15
  }
  __syncthreads();

  if (tid < 64) {
    const int rg2 = tid >> 4;
    const int lr  = tid & 15;
    const float y = ybuf[rg2][lr] + ybuf[rg2 + 4][lr] + b2[h];
    const int n = nb + rg2 * 16 + lr;
    out[(size_t)n * OUT_ROW + h * (D_IN + 1) + D_IN] = y;
  }
}

// Pure-bandwidth kernel: out[n, h*513 + 0:512] = x[n, :] for all 32 heads.
// (513-float stride breaks 16B alignment -> scalar b32 stores, fully coalesced.)
__global__ __launch_bounds__(256)
void bcast_x(const float* __restrict__ x, float* __restrict__ out) {
  const int n = blockIdx.x;
  const int t = threadIdx.x;            // 256 threads
  const float a = x[(size_t)n * D_IN + t];
  const float b = x[(size_t)n * D_IN + t + 256];
  float* orow = out + (size_t)n * OUT_ROW;
#pragma unroll
  for (int h = 0; h < USED; ++h) {
    orow[h * (D_IN + 1) + t]       = a;
    orow[h * (D_IN + 1) + 256 + t] = b;
  }
}

extern "C" void kernel_launch(void* const* d_in, const int* in_sizes, int n_in,
                              void* d_out, int out_size, void* d_ws, size_t ws_size,
                              hipStream_t stream) {
  const float* x  = (const float*)d_in[0];
  const float* W1 = (const float*)d_in[1];
  const float* b1 = (const float*)d_in[2];
  const float* W2 = (const float*)d_in[3];
  const float* b2 = (const float*)d_in[4];
  float* out = (float*)d_out;
  (void)in_sizes; (void)n_in; (void)out_size; (void)d_ws; (void)ws_size;

  fused_head_gemm<<<dim3(N_ROWS / 64, USED), 256, 0, stream>>>(x, W1, b1, W2, b2, out);
  bcast_x<<<dim3(N_ROWS), 256, 0, stream>>>(x, out);
}
